// RNLayer_23545010717283
// MI455X (gfx1250) — compile-verified, matrix operand order checked by reading, not run on hardware
//
#include <hip/hip_runtime.h>
#include <hip/hip_bf16.h>

typedef _Float16 v16h __attribute__((ext_vector_type(16)));
typedef _Float16 v8h  __attribute__((ext_vector_type(8)));
typedef float    v8f  __attribute__((ext_vector_type(8)));

#define NBATCH 2048
#define NCH    24          // conv channels
#define NPAIR  625         // 25*25 pairs per batch
#define QDIM   10
#define HDIM   256
#define NTL    4           // M-tiles per wave (4 wmma per B fragment)
#define TILE_HALF (16 * HDIM)   // halfs per 16-row tile region (8KB)

// f16 weight-fragment workspace (halfs). Per (kt,nt) tile: 512 halfs, lane-major.
// Element i of lane's v16h B-fragment holds W[k][n] with
//   k = kt*32 + (lane>=16 ? 8 : 0) + (i&7) + (i>=8 ? 16 : 0),  n = nt*16 + (lane&15)
#define W1_BASE 0           // 2 kt * 16 nt tiles (K padded 62->64)
#define W2_BASE 16384
#define W3_BASE 81920
#define W4_BASE 147456
#define WS_TOTAL 212992     // halfs

__global__ void prep_weights(const float* __restrict__ W1, const float* __restrict__ W2,
                             const float* __restrict__ W3, const float* __restrict__ W4,
                             _Float16* __restrict__ ws) {
    int t = blockIdx.x * 256 + threadIdx.x;
    if (t >= WS_TOTAL) return;
    const float* W; int Kdim; int rem;
    if (t < W2_BASE)      { W = W1; Kdim = 62;  rem = t; }
    else if (t < W3_BASE) { W = W2; Kdim = 256; rem = t - W2_BASE; }
    else if (t < W4_BASE) { W = W3; Kdim = 256; rem = t - W3_BASE; }
    else                  { W = W4; Kdim = 256; rem = t - W4_BASE; }
    int tile = rem >> 9;
    int lane = (rem >> 4) & 31;
    int i    = rem & 15;
    int kt = tile >> 4, nt = tile & 15;
    int k = kt * 32 + ((lane >> 4) ? 8 : 0) + (i & 7) + ((i >= 8) ? 16 : 0);
    int n = nt * 16 + (lane & 15);
    float v = (k < Kdim) ? W[k * HDIM + n] : 0.0f;
    ws[t] = (_Float16)v;
}

// Zero-cost compiler memory barrier: LDS scratch is wave-private and the LDS
// pipe is in-order per wave, so we only need to stop compiler reordering.
__device__ __forceinline__ void cbar() { asm volatile("" ::: "memory"); }

// A-fragment: 16x32 f16 tile from row-major LDS (stride halfs).
__device__ __forceinline__ v16h load_afrag(const _Float16* base, int stride, int lane, int kt) {
    int row = lane & 15;
    int kb  = kt * 32 + ((lane >> 4) ? 8 : 0);
    const _Float16* p = base + row * stride + kb;
    v8h lo = *(const v8h*)(p);
    v8h hi = *(const v8h*)(p + 16);
    return __builtin_shufflevector(lo, hi, 0,1,2,3,4,5,6,7,8,9,10,11,12,13,14,15);
}

// B-fragment: 32-byte global load; opaque address defeats LICM/CSE (no spills).
__device__ __forceinline__ v16h load_bfrag(const _Float16* p) {
    const __attribute__((address_space(1))) v16h* gp =
        (const __attribute__((address_space(1))) v16h*)p;
    asm volatile("" : "+v"(gp));
    return *gp;
}

__device__ __forceinline__ v8f wmma16(v16h a, v16h b, v8f c) {
    return __builtin_amdgcn_wmma_f32_16x16x32_f16(false, a, false, b, (short)0, c, false, false);
}

// One layer over NTL 16-row tiles sharing each B fragment, with one-deep
// fragment prefetch pipelined across the whole (nt,kt) sequence.
template<int NKT, bool FINAL>
__device__ __forceinline__ void rn_layer(const _Float16* __restrict__ ws, int wbase,
                                         const float* __restrict__ bias,
                                         _Float16* my, int strideIn, int lane,
                                         int t0, float* regOut) {
    v16h a[NTL][NKT];
    #pragma unroll
    for (int tt = 0; tt < NTL; ++tt)
        #pragma unroll
        for (int kt = 0; kt < NKT; ++kt)
            a[tt][kt] = load_afrag(my + tt * TILE_HALF, strideIn, lane, kt);
    cbar();   // fragments in regs; stores below may overwrite scratch

    const int col   = lane & 15;
    const int rbase = (lane >> 4) * 8;
    #define BADDR(kt_, nt_) (ws + wbase + (((kt_) * 16 + (nt_)) * 512) + lane * 16)

    v16h bf = load_bfrag(BADDR(0, 0));
    #pragma unroll
    for (int nt = 0; nt < 16; ++nt) {
        float bi = bias[nt * 16 + col];
        v8f acc[NTL];
        #pragma unroll
        for (int tt = 0; tt < NTL; ++tt) acc[tt] = (v8f){};
        #pragma unroll
        for (int kt = 0; kt < NKT; ++kt) {
            v16h cur = bf;
            int kn = kt + 1, nn = nt;
            if (kn == NKT) { kn = 0; nn = nt + 1; }
            if (nn < 16) bf = load_bfrag(BADDR(kn, nn));   // prefetch next fragment
            #pragma unroll
            for (int tt = 0; tt < NTL; ++tt)
                acc[tt] = wmma16(a[tt][kt], cur, acc[tt]);
        }
        int n = nt * 16 + col;
        if (!FINAL) {
            #pragma unroll
            for (int tt = 0; tt < NTL; ++tt)
                #pragma unroll
                for (int r = 0; r < 8; ++r) {
                    float v = acc[tt][r] + bi;
                    v = v > 0.0f ? v : 0.0f;
                    my[tt * TILE_HALF + (rbase + r) * HDIM + n] = (_Float16)v;
                }
        } else {
            float s = 0.0f;
            #pragma unroll
            for (int tt = 0; tt < NTL; ++tt)
                #pragma unroll
                for (int r = 0; r < 8; ++r) {
                    int p = (t0 + tt) * 16 + rbase + r;
                    float v = acc[tt][r] + bi;
                    v = v > 0.0f ? v : 0.0f;
                    if (p < NPAIR) s += v;
                }
            regOut[nt] += s;
        }
    }
    #undef BADDR
    cbar();
}

__global__ __launch_bounds__(64) void rn_main(const float* __restrict__ x,
                                              const float* __restrict__ ques,
                                              const float* __restrict__ b1,
                                              const float* __restrict__ b2,
                                              const float* __restrict__ b3,
                                              const float* __restrict__ b4,
                                              const _Float16* __restrict__ ws,
                                              float* __restrict__ out) {
    __shared__ _Float16 scr[2 * NTL * TILE_HALF];   // 64KB: 2 waves x 4 tiles x 8KB

    const int tid  = threadIdx.x;
    const int lane = tid & 31;
    const int wave = tid >> 5;
    const int b    = blockIdx.x;

    _Float16* my = scr + wave * (NTL * TILE_HALF);
    const int col   = lane & 15;
    const int rbase = (lane >> 4) * 8;

    float regOut[16];
    #pragma unroll
    for (int nt = 0; nt < 16; ++nt) regOut[nt] = 0.0f;

    for (int pp = 0; pp < 5; ++pp) {
        const int t0 = (pp * 2 + wave) * NTL;     // tiles t0 .. t0+3 (0..39)

        // ---- synthesize features: two 62-col rows per lane (f16, stride 64) ----
        #pragma unroll
        for (int half = 0; half < 2; ++half) {
            int rr = lane + half * 32;            // row within this wave's 64 rows
            int p  = t0 * 16 + rr;                // global pair index
            _Float16* dst = my + (rr >> 4) * TILE_HALF + (rr & 15) * 64;
            if (p >= NPAIR) {
                #pragma unroll 8
                for (int c = 0; c < 64; ++c) dst[c] = (_Float16)0.0f;
            } else {
                int oi = p % 25, oj = p / 25;     // feat = [x_i | x_j | ques]
                for (int c = 0; c < 64; ++c) {
                    float v;
                    if (c < 52) {
                        int obj = (c < 26) ? oi : oj;
                        int cc  = (c < 26) ? c  : (c - 26);
                        if (cc < 24)       v = x[(b * NCH + cc) * 25 + obj];
                        else if (cc == 24) v = -2.5f + 1.25f * (float)(obj % 5); // cx
                        else               v = -2.5f + 1.25f * (float)(obj / 5); // cy
                    } else if (c < 62) {
                        v = ques[b * QDIM + (c - 52)];
                    } else {
                        v = 0.0f;
                    }
                    dst[c] = (_Float16)v;
                }
            }
        }
        cbar();

        rn_layer<2, false>(ws, W1_BASE, b1, my, 64,   lane, t0, regOut); // 62->256
        rn_layer<8, false>(ws, W2_BASE, b2, my, HDIM, lane, t0, regOut); // 256->256
        rn_layer<8, false>(ws, W3_BASE, b3, my, HDIM, lane, t0, regOut); // 256->256
        rn_layer<8, true >(ws, W4_BASE, b4, my, HDIM, lane, t0, regOut); // 256->sum
    }

    // ---- cross-wave reduction through (now free) scratch ----
    float* fscr = (float*)scr;   // 2 waves * 2 halves * 256 floats
    __syncthreads();             // all waves done with their scratch
    #pragma unroll
    for (int nt = 0; nt < 16; ++nt)
        fscr[wave * 512 + (lane >> 4) * 256 + nt * 16 + col] = regOut[nt];
    __syncthreads();
    for (int t = tid; t < HDIM; t += 64) {
        float s = 0.0f;
        #pragma unroll
        for (int w = 0; w < 2; ++w)
            s += fscr[w * 512 + t] + fscr[w * 512 + 256 + t];
        out[b * HDIM + t] = s;
    }
}

extern "C" void kernel_launch(void* const* d_in, const int* in_sizes, int n_in,
                              void* d_out, int out_size, void* d_ws, size_t ws_size,
                              hipStream_t stream) {
    const float* x    = (const float*)d_in[0];
    const float* ques = (const float*)d_in[1];
    const float* W1   = (const float*)d_in[2];
    const float* b1   = (const float*)d_in[3];
    const float* W2   = (const float*)d_in[4];
    const float* b2   = (const float*)d_in[5];
    const float* W3   = (const float*)d_in[6];
    const float* b3   = (const float*)d_in[7];
    const float* W4   = (const float*)d_in[8];
    const float* b4   = (const float*)d_in[9];
    float*    out = (float*)d_out;
    _Float16* wsH = (_Float16*)d_ws;

    prep_weights<<<WS_TOTAL / 256, 256, 0, stream>>>(W1, W2, W3, W4, wsH);
    rn_main<<<NBATCH, 64, 0, stream>>>(x, ques, b1, b2, b3, b4, wsH, out);
}
